// LunaDecoderLayer_25391846654054
// MI455X (gfx1250) — matrix operand order repaired and checked by reading, not compiled, not run on hardware
//
#include <hip/hip_runtime.h>
#include <math.h>

typedef __bf16 bf16;
typedef __attribute__((ext_vector_type(16))) bf16 bf16x16;
typedef __attribute__((ext_vector_type(8)))  bf16 bf16x8;
typedef __attribute__((ext_vector_type(8)))  float f32x8;

#define N_SEQ 2048
#define BSZ   2
#define EMB   1024
#define NHEAD 16
#define HD    64
#define PLEN  16
#define M_TOK (N_SEQ*BSZ)   // 4096 tokens, row m = b*N_SEQ + i
#define BH    (BSZ*NHEAD)   // 32 head-batches
#define LC    64            // scan chunk length
#define NCH   (N_SEQ/LC)    // 32 chunks per sequence
#define SCAL  0.125f        // hd^-0.5

union FragAB { bf16x16 v; bf16x8 h[2]; };

__device__ __forceinline__ f32x8 wmma_bf16(const FragAB& a, const FragAB& b, f32x8 c) {
  return __builtin_amdgcn_wmma_f32_16x16x32_bf16(false, a.v, false, b.v, (short)0, c, false, false);
}

// ---------------------------------------------------------------------------
// C(M,N) = alpha * (A(M,K) @ W(N,K)^T + bias)   [bf16 in, fp32 accum]
// Per wave: 32x32 output (2x2 tiles of 16x16), 8 waves/block -> 64x128 tile.
// ---------------------------------------------------------------------------
__global__ __launch_bounds__(256) void luna_gemm_bf16(
    const bf16* __restrict__ A, const bf16* __restrict__ W,
    const float* __restrict__ bias, float alpha,
    bf16* __restrict__ Cbf, float* __restrict__ Cf, int permuteOut,
    int M, int N, int K)
{
  const int lane = threadIdx.x & 31;
  const int wave = threadIdx.x >> 5;
  const int kh   = lane >> 4;   // half-wave id (0/1)
  const int l16  = lane & 15;
  const int m_base = blockIdx.y * 64  + (wave >> 2) * 32;
  const int n_base = blockIdx.x * 128 + (wave & 3)  * 32;

  f32x8 acc[2][2] = {};

  const bf16* Ar0 = A + (size_t)(m_base + l16) * K;
  const bf16* Ar1 = Ar0 + (size_t)16 * K;
  const bf16* Wr0 = W + (size_t)(n_base + l16) * K;
  const bf16* Wr1 = Wr0 + (size_t)16 * K;

  for (int k0 = 0; k0 < K; k0 += 32) {
    FragAB a0, a1, b0, b1;
    a0.h[0] = *(const bf16x8*)(Ar0 + k0 + kh * 16);
    a0.h[1] = *(const bf16x8*)(Ar0 + k0 + kh * 16 + 8);
    a1.h[0] = *(const bf16x8*)(Ar1 + k0 + kh * 16);
    a1.h[1] = *(const bf16x8*)(Ar1 + k0 + kh * 16 + 8);
    b0.h[0] = *(const bf16x8*)(Wr0 + k0 + kh * 16);
    b0.h[1] = *(const bf16x8*)(Wr0 + k0 + kh * 16 + 8);
    b1.h[0] = *(const bf16x8*)(Wr1 + k0 + kh * 16);
    b1.h[1] = *(const bf16x8*)(Wr1 + k0 + kh * 16 + 8);
    acc[0][0] = wmma_bf16(a0, b0, acc[0][0]);
    acc[0][1] = wmma_bf16(a0, b1, acc[0][1]);
    acc[1][0] = wmma_bf16(a1, b0, acc[1][0]);
    acc[1][1] = wmma_bf16(a1, b1, acc[1][1]);
  }

  #pragma unroll
  for (int i = 0; i < 2; i++) {
    #pragma unroll
    for (int j = 0; j < 2; j++) {
      #pragma unroll
      for (int v = 0; v < 8; v++) {
        const int row = m_base + i * 16 + v + kh * 8;
        const int col = n_base + j * 16 + l16;
        const float val = alpha * (acc[i][j][v] + bias[col]);
        if (Cbf) {
          Cbf[(size_t)row * N + col] = (bf16)val;
        } else {
          const int pr = permuteOut ? ((row % N_SEQ) * BSZ + (row / N_SEQ)) : row;
          Cf[(size_t)pr * N + col] = val;
        }
      }
    }
  }
}

// query (n,B,E) fp32 -> bf16 token-major rows m = b*n + i
__global__ void luna_cvt_query(const float* __restrict__ q, bf16* __restrict__ dst) {
  int t = blockIdx.x * blockDim.x + threadIdx.x;
  if (t >= N_SEQ * BSZ * EMB) return;
  int e = t % EMB; int r = t / EMB; int b = r % BSZ; int i = r / BSZ;
  dst[(size_t)(b * N_SEQ + i) * EMB + e] = (bf16)q[t];
}

__global__ void luna_cvt_f32_bf16(const float* __restrict__ src, bf16* __restrict__ dst, int count) {
  int t = blockIdx.x * blockDim.x + threadIdx.x;
  if (t < count) dst[t] = (bf16)src[t];
}

// pq[bh][p][d] = SCAL * (pquery[p,b,:] . W_pq[h*64+d,:] + b_pq[h*64+d])
__global__ void luna_pq(const float* __restrict__ pquery, const float* __restrict__ W_pq,
                        const float* __restrict__ b_pq, float* __restrict__ pq_f) {
  int t = blockIdx.x * blockDim.x + threadIdx.x;  // BH*PLEN*HD = 32768
  if (t >= BH * PLEN * HD) return;
  int d = t % HD; int p = (t / HD) % PLEN; int bh = t / (HD * PLEN);
  int b = bh / NHEAD, h = bh % NHEAD;
  int row = h * HD + d;
  const float* x = pquery + (size_t)(p * BSZ + b) * EMB;
  const float* w = W_pq + (size_t)row * EMB;
  float acc = b_pq[row];
  for (int e = 0; e < EMB; e++) acc += x[e] * w[e];
  pq_f[(size_t)bh * (PLEN * HD) + p * HD + d] = SCAL * acc;
}

// pattn[bh][i][p] = softplus_{beta=ln2}(k_p . pq) = log2(1 + 2^x)
__global__ void luna_pattn(const bf16* __restrict__ kp_bf, const float* __restrict__ pq_f,
                           float* __restrict__ pattn_f) {
  int t = blockIdx.x * blockDim.x + threadIdx.x;  // BH*N_SEQ*PLEN = 1M
  if (t >= BH * N_SEQ * PLEN) return;
  int p = t % PLEN; int i = (t / PLEN) % N_SEQ; int bh = t / (PLEN * N_SEQ);
  int b = bh / NHEAD, h = bh % NHEAD;
  const bf16* kp = kp_bf + (size_t)(b * N_SEQ + i) * EMB + h * HD;
  const float* pqv = pq_f + (size_t)bh * (PLEN * HD) + p * HD;
  float acc = 0.f;
  #pragma unroll 8
  for (int d = 0; d < HD; d++) acc += (float)kp[d] * pqv[d];
  pattn_f[t] = (acc > 15.f) ? acc : log2f(1.f + exp2f(acc));
}

// pattn packs: p-major transpose (WMMA B over token-K) and padded rows (WMMA B over p-K)
__global__ void luna_pack_pattn(const float* __restrict__ pattn_f,
                                bf16* __restrict__ pattnT_bf, bf16* __restrict__ pattnP_bf) {
  int t = blockIdx.x * blockDim.x + threadIdx.x;
  if (t >= BH * N_SEQ * PLEN) return;
  int p = t % PLEN; int i = (t / PLEN) % N_SEQ; int bh = t / (PLEN * N_SEQ);
  bf16 v = (bf16)pattn_f[t];
  pattnT_bf[(size_t)(bh * PLEN + p) * N_SEQ + i] = v;
  pattnP_bf[(size_t)(bh * N_SEQ + i) * 32 + p] = v;
  pattnP_bf[(size_t)(bh * N_SEQ + i) * 32 + 16 + p] = (bf16)0.f;
}

// kv transpose: kvT[bh][d][i]  (WMMA B for Out = B2 @ KV, K over tokens)
__global__ void luna_pack_kvT(const bf16* __restrict__ kv_bf, bf16* __restrict__ kvT_bf) {
  int t = blockIdx.x * blockDim.x + threadIdx.x;  // BH*N_SEQ*HD = 4M
  if (t >= BH * N_SEQ * HD) return;
  int d = t % HD; int i = (t / HD) % N_SEQ; int bh = t / (HD * N_SEQ);
  int b = bh / NHEAD, h = bh % NHEAD;
  kvT_bf[(size_t)(bh * HD + d) * N_SEQ + i] =
      kv_bf[(size_t)(b * N_SEQ + i) * EMB + h * HD + d];
}

// Pass 1: per-(bh,chunk) partial sums  csum[bh][ch][d][p]
__global__ __launch_bounds__(256) void luna_chunk_sums(
    const bf16* __restrict__ kv_bf, const float* __restrict__ pattn_f,
    float* __restrict__ csum) {
  const int blk = blockIdx.x;            // bh*NCH + ch
  const int bh = blk / NCH, ch = blk % NCH;
  const int b = bh / NHEAD, h = bh % NHEAD;
  const int d = threadIdx.x & 63, pg = threadIdx.x >> 6;
  __shared__ float sp[PLEN];
  float acc[4] = {0.f, 0.f, 0.f, 0.f};
  for (int ii = 0; ii < LC; ii++) {
    const int i = ch * LC + ii;
    if (threadIdx.x < PLEN)
      sp[threadIdx.x] = pattn_f[((size_t)bh * N_SEQ + i) * PLEN + threadIdx.x];
    __syncthreads();
    const float kvd = (float)kv_bf[(size_t)(b * N_SEQ + i) * EMB + h * HD + d];
    #pragma unroll
    for (int j = 0; j < 4; j++) acc[j] += kvd * sp[pg * 4 + j];
    __syncthreads();
  }
  float* o = csum + (((size_t)blk * HD + d) * PLEN + pg * 4);
  #pragma unroll
  for (int j = 0; j < 4; j++) o[j] = acc[j];
}

// Exclusive prefix over chunks; emits S_base in both WMMA-B-ready bf16 forms:
// sbaseT[blk][p][d] (K over d) and sbaseP[blk][d][32] padded (K over p)
__global__ __launch_bounds__(256) void luna_prefix(const float* __restrict__ csum,
                                                   bf16* __restrict__ sbaseT_bf,
                                                   bf16* __restrict__ sbaseP_bf) {
  const int bh = blockIdx.x;
  float run[4] = {0.f, 0.f, 0.f, 0.f};
  for (int ch = 0; ch < NCH; ch++) {
    const int blk = bh * NCH + ch;
    #pragma unroll
    for (int j = 0; j < 4; j++) {
      const int o = threadIdx.x + 256 * j;  // = d*16 + p
      const int d = o >> 4, p = o & 15;
      bf16 v = (bf16)run[j];
      sbaseT_bf[(size_t)(blk * PLEN + p) * HD + d] = v;
      sbaseP_bf[(size_t)(blk * HD + d) * 32 + p] = v;
      sbaseP_bf[(size_t)(blk * HD + d) * 32 + 16 + p] = (bf16)0.f;
      run[j] += csum[(size_t)blk * (HD * PLEN) + o];
    }
  }
}

// ---------------------------------------------------------------------------
// Pass 2 (WMMA chunked linear attention). One block (4 waves) per (bh,chunk);
// wave r owns 16-row strip r.
//   W   = Q@Sbase + causal(Q@KV^T)@Pattn        -> /(i+1), softmax(16)
//   Out = P@Sbase^T + causal(P@Pattn^T)@KV      -> /(i+1)
// Intermediates round-trip per-wave LDS strips to re-enter A-fragment layout.
// ---------------------------------------------------------------------------
__global__ __launch_bounds__(128) void luna_scan_wmma(
    const bf16* __restrict__ qs_bf, const bf16* __restrict__ kv_bf,
    const bf16* __restrict__ kvT_bf, const bf16* __restrict__ pattnT_bf,
    const bf16* __restrict__ pattnP_bf, const bf16* __restrict__ sbaseT_bf,
    const bf16* __restrict__ sbaseP_bf, bf16* __restrict__ attn_bf)
{
  const int blk = blockIdx.x;          // bh*NCH + ch
  const int bh = blk / NCH, ch = blk % NCH;
  const int b = bh / NHEAD, h = bh % NHEAD;
  const int lane = threadIdx.x & 31;
  const int r = __builtin_amdgcn_readfirstlane(threadIdx.x >> 5);  // strip id, scalar
  const int kh = lane >> 4, l16 = lane & 15;

  __shared__ __align__(16) bf16 ldsA[4][16 * 64];  // per-wave strip (QK^T, then B2)
  __shared__ __align__(16) bf16 ldsP[4][16 * 32];  // per-wave probs, K-padded

  // ---- Stage 1: W_strip ----
  FragAB qA[2];
  {
    const bf16* qrow = qs_bf + (size_t)(b * N_SEQ + ch * LC + r * 16 + l16) * EMB + h * HD;
    #pragma unroll
    for (int ks = 0; ks < 2; ks++) {
      qA[ks].h[0] = *(const bf16x8*)(qrow + ks * 32 + kh * 16);
      qA[ks].h[1] = *(const bf16x8*)(qrow + ks * 32 + kh * 16 + 8);
    }
  }
  f32x8 acc = {};
  {
    const bf16* srow = sbaseT_bf + (size_t)(blk * PLEN + l16) * HD;
    #pragma unroll
    for (int ks = 0; ks < 2; ks++) {
      FragAB sb;
      sb.h[0] = *(const bf16x8*)(srow + ks * 32 + kh * 16);
      sb.h[1] = *(const bf16x8*)(srow + ks * 32 + kh * 16 + 8);
      acc = wmma_bf16(qA[ks], sb, acc);
    }
  }
  for (int c = 0; c < 4; c++) {
    f32x8 qk = {};
    if (c <= r) {  // scalar branch (r uniform per wave): EXEC stays all-1 for WMMA
      const bf16* krow = kv_bf + (size_t)(b * N_SEQ + ch * LC + c * 16 + l16) * EMB + h * HD;
      #pragma unroll
      for (int ks = 0; ks < 2; ks++) {
        FragAB kb;
        kb.h[0] = *(const bf16x8*)(krow + ks * 32 + kh * 16);
        kb.h[1] = *(const bf16x8*)(krow + ks * 32 + kh * 16 + 8);
        qk = wmma_bf16(qA[ks], kb, qk);
      }
      if (c == r) {
        #pragma unroll
        for (int v = 0; v < 8; v++) qk[v] = (l16 <= v + 8 * kh) ? qk[v] : 0.f;
      }
    }
    #pragma unroll
    for (int v = 0; v < 8; v++)
      ldsA[r][(v + 8 * kh) * 64 + c * 16 + l16] = (bf16)qk[v];
  }
  __syncthreads();
  #pragma unroll
  for (int ks = 0; ks < 2; ks++) {
    FragAB aF, pB;
    aF.h[0] = *(const bf16x8*)(&ldsA[r][l16 * 64 + ks * 32 + kh * 16]);
    aF.h[1] = *(const bf16x8*)(&ldsA[r][l16 * 64 + ks * 32 + kh * 16 + 8]);
    const bf16* pr = pattnT_bf + (size_t)(bh * PLEN + l16) * N_SEQ + ch * LC;
    pB.h[0] = *(const bf16x8*)(pr + ks * 32 + kh * 16);
    pB.h[1] = *(const bf16x8*)(pr + ks * 32 + kh * 16 + 8);
    acc = wmma_bf16(aF, pB, acc);
  }

  // ---- /(i+1) + softmax over 16 cols (row lives on a 16-lane group) ----
  float inv[8], prob[8];
  #pragma unroll
  for (int v = 0; v < 8; v++) {
    const int i = ch * LC + r * 16 + v + 8 * kh;
    inv[v] = 1.f / (float)(i + 1);
    float wv = acc[v] * inv[v];
    float mx = wv;
    #pragma unroll
    for (int msk = 1; msk < 16; msk <<= 1) mx = fmaxf(mx, __shfl_xor(mx, msk, 32));
    float e = expf(wv - mx);
    float se = e;
    #pragma unroll
    for (int msk = 1; msk < 16; msk <<= 1) se += __shfl_xor(se, msk, 32);
    prob[v] = e / se;
  }

  // ---- Stage 2: Out_strip ----
  #pragma unroll
  for (int v = 0; v < 8; v++) {
    ldsP[r][(v + 8 * kh) * 32 + l16] = (bf16)prob[v];
    ldsP[r][(v + 8 * kh) * 32 + 16 + l16] = (bf16)0.f;
  }
  __syncthreads();
  FragAB pA;
  pA.h[0] = *(const bf16x8*)(&ldsP[r][l16 * 32 + kh * 16]);
  pA.h[1] = *(const bf16x8*)(&ldsP[r][l16 * 32 + kh * 16 + 8]);

  f32x8 oacc[4];
  #pragma unroll
  for (int nt = 0; nt < 4; nt++) {
    f32x8 z = {};
    FragAB sb;
    const bf16* srow = sbaseP_bf + (size_t)(blk * HD + nt * 16 + l16) * 32;
    sb.h[0] = *(const bf16x8*)(srow + kh * 16);
    sb.h[1] = *(const bf16x8*)(srow + kh * 16 + 8);
    oacc[nt] = wmma_bf16(pA, sb, z);
  }
  for (int c = 0; c < 4; c++) {
    f32x8 b2 = {};
    if (c <= r) {
      FragAB pp;
      const bf16* prow = pattnP_bf + (size_t)(bh * N_SEQ + ch * LC + c * 16 + l16) * 32;
      pp.h[0] = *(const bf16x8*)(prow + kh * 16);
      pp.h[1] = *(const bf16x8*)(prow + kh * 16 + 8);
      b2 = wmma_bf16(pA, pp, b2);
      if (c == r) {
        #pragma unroll
        for (int v = 0; v < 8; v++) b2[v] = (l16 <= v + 8 * kh) ? b2[v] : 0.f;
      }
    }
    #pragma unroll
    for (int v = 0; v < 8; v++)
      ldsA[r][(v + 8 * kh) * 64 + c * 16 + l16] = (bf16)b2[v];
  }
  __syncthreads();
  FragAB b2A[2];
  #pragma unroll
  for (int ks = 0; ks < 2; ks++) {
    b2A[ks].h[0] = *(const bf16x8*)(&ldsA[r][l16 * 64 + ks * 32 + kh * 16]);
    b2A[ks].h[1] = *(const bf16x8*)(&ldsA[r][l16 * 64 + ks * 32 + kh * 16 + 8]);
  }
  #pragma unroll
  for (int nt = 0; nt < 4; nt++) {
    #pragma unroll
    for (int ks = 0; ks < 2; ks++) {
      FragAB kb;
      const bf16* krow = kvT_bf + (size_t)(bh * HD + nt * 16 + l16) * N_SEQ + ch * LC;
      kb.h[0] = *(const bf16x8*)(krow + ks * 32 + kh * 16);
      kb.h[1] = *(const bf16x8*)(krow + ks * 32 + kh * 16 + 8);
      oacc[nt] = wmma_bf16(b2A[ks], kb, oacc[nt]);
    }
  }
  #pragma unroll
  for (int nt = 0; nt < 4; nt++) {
    #pragma unroll
    for (int v = 0; v < 8; v++) {
      const int i = ch * LC + r * 16 + v + 8 * kh;
      attn_bf[(size_t)(b * N_SEQ + i) * EMB + h * HD + nt * 16 + l16] =
          (bf16)(oacc[nt][v] * inv[v]);
    }
  }
}

// ---------------------------------------------------------------------------
extern "C" void kernel_launch(void* const* d_in, const int* in_sizes, int n_in,
                              void* d_out, int out_size, void* d_ws, size_t ws_size,
                              hipStream_t stream) {
  const float* query  = (const float*)d_in[0];
  const float* pquery = (const float*)d_in[1];
  const float* W_pq   = (const float*)d_in[2];
  const float* b_pq   = (const float*)d_in[3];
  const float* W_q    = (const float*)d_in[4];
  const float* b_q    = (const float*)d_in[5];
  const float* W_pc   = (const float*)d_in[6];
  const float* b_pc   = (const float*)d_in[7];
  const float* W_c    = (const float*)d_in[8];
  const float* b_c    = (const float*)d_in[9];
  const float* W_out  = (const float*)d_in[10];
  const float* b_out  = (const float*)d_in[11];
  float* out = (float*)d_out;

  char* wp = (char*)d_ws;
  auto take = [&](size_t bytes) { void* r = (void*)wp; wp += (bytes + 255) & ~(size_t)255; return r; };
  bf16*  q_in     = (bf16*)take((size_t)M_TOK * EMB * 2);
  bf16*  wq_bf    = (bf16*)take((size_t)EMB * EMB * 2);
  bf16*  wpc_bf   = (bf16*)take((size_t)EMB * EMB * 2);
  bf16*  wc_bf    = (bf16*)take((size_t)EMB * EMB * 2);
  bf16*  wout_bf  = (bf16*)take((size_t)EMB * EMB * 2);
  bf16*  kp_bf    = (bf16*)take((size_t)M_TOK * EMB * 2);
  bf16*  qs_bf    = (bf16*)take((size_t)M_TOK * EMB * 2);
  bf16*  kv_bf    = (bf16*)take((size_t)M_TOK * EMB * 2);
  bf16*  kvT_bf   = (bf16*)take((size_t)BH * HD * N_SEQ * 2);
  float* pq_f     = (float*)take((size_t)BH * PLEN * HD * 4);
  float* pattn_f  = (float*)take((size_t)BH * N_SEQ * PLEN * 4);
  bf16*  pattnT_bf= (bf16*)take((size_t)BH * PLEN * N_SEQ * 2);
  bf16*  pattnP_bf= (bf16*)take((size_t)BH * N_SEQ * 32 * 2);
  float* csum     = (float*)take((size_t)BH * NCH * HD * PLEN * 4);
  bf16*  sbaseT_bf= (bf16*)take((size_t)BH * NCH * PLEN * HD * 2);
  bf16*  sbaseP_bf= (bf16*)take((size_t)BH * NCH * HD * 32 * 2);
  bf16*  attn_bf  = (bf16*)take((size_t)M_TOK * EMB * 2);
  (void)ws_size; (void)n_in; (void)in_sizes; (void)out_size;

  const int NE = N_SEQ * BSZ * EMB;   // 4,194,304
  const int WE = EMB * EMB;           // 1,048,576

  // 1) convert activations & weights to bf16
  luna_cvt_query<<<(NE + 255) / 256, 256, 0, stream>>>(query, q_in);
  luna_cvt_f32_bf16<<<(WE + 255) / 256, 256, 0, stream>>>(W_q,   wq_bf,   WE);
  luna_cvt_f32_bf16<<<(WE + 255) / 256, 256, 0, stream>>>(W_pc,  wpc_bf,  WE);
  luna_cvt_f32_bf16<<<(WE + 255) / 256, 256, 0, stream>>>(W_c,   wc_bf,   WE);
  luna_cvt_f32_bf16<<<(WE + 255) / 256, 256, 0, stream>>>(W_out, wout_bf, WE);

  // 2) three input projections (WMMA)
  dim3 ggrid(EMB / 128, M_TOK / 64);
  luna_gemm_bf16<<<ggrid, 256, 0, stream>>>(q_in, wpc_bf, b_pc, 1.0f,
                                            kp_bf, nullptr, 0, M_TOK, EMB, EMB);
  luna_gemm_bf16<<<ggrid, 256, 0, stream>>>(q_in, wq_bf, b_q, SCAL,
                                            qs_bf, nullptr, 0, M_TOK, EMB, EMB);
  luna_gemm_bf16<<<ggrid, 256, 0, stream>>>(q_in, wc_bf, b_c, 1.0f,
                                            kv_bf, nullptr, 0, M_TOK, EMB, EMB);

  // 3) pq projection (tiny) + pattn (softplus) + packs
  luna_pq<<<(BH * PLEN * HD + 255) / 256, 256, 0, stream>>>(pquery, W_pq, b_pq, pq_f);
  luna_pattn<<<(BH * N_SEQ * PLEN + 255) / 256, 256, 0, stream>>>(kp_bf, pq_f, pattn_f);
  luna_pack_pattn<<<(BH * N_SEQ * PLEN + 255) / 256, 256, 0, stream>>>(pattn_f, pattnT_bf, pattnP_bf);
  luna_pack_kvT<<<(BH * N_SEQ * HD + 255) / 256, 256, 0, stream>>>(kv_bf, kvT_bf);

  // 4) chunked causal scan: partial sums -> exclusive prefix -> WMMA chunk pass
  luna_chunk_sums<<<BH * NCH, 256, 0, stream>>>(kv_bf, pattn_f, csum);
  luna_prefix<<<BH, 256, 0, stream>>>(csum, sbaseT_bf, sbaseP_bf);
  luna_scan_wmma<<<BH * NCH, 128, 0, stream>>>(qs_bf, kv_bf, kvT_bf, pattnT_bf,
                                               pattnP_bf, sbaseT_bf, sbaseP_bf, attn_bf);

  // 5) output projection (WMMA), permute rows (b*n+i) -> (i*B+b), fp32 out
  luna_gemm_bf16<<<ggrid, 256, 0, stream>>>(attn_bf, wout_bf, b_out, 1.0f,
                                            nullptr, out, 1, M_TOK, EMB, EMB);
}
